// MemoryEfficientAttnBlock_17162689315175
// MI455X (gfx1250) — matrix operand order repaired and checked by reading, not compiled, not run on hardware
//
#include <hip/hip_runtime.h>
#include <hip/hip_bf16.h>

typedef _Float16 half_t;
typedef __attribute__((ext_vector_type(16))) _Float16 v16h;
typedef __attribute__((ext_vector_type(8)))  float    v8f;

// ---------------------------------------------------------------------------
// WMMA helper (CDNA5 wave32, V_WMMA_F32_16X16X32_F16)
// ---------------------------------------------------------------------------
__device__ __forceinline__ v8f wmma_f16(v16h a, v16h b, v8f c) {
  return __builtin_amdgcn_wmma_f32_16x16x32_f16(false, a, false, b, (short)0, c,
                                                false, false);
}

// A-matrix 16x32 f16 fragment from row-major src (stride lda halves).
// lanes 0-15 -> M=0..15 hold K {0..7,16..23}; lanes 16-31 hold K {8..15,24..31}.
// Two contiguous 8-half (16B) reads -> 2x ds_load_b128.
__device__ __forceinline__ v16h load_a(const half_t* __restrict__ src, int lda,
                                       int lane) {
  int m  = lane & 15;
  int kh = (lane < 16) ? 0 : 8;
  const half_t* p = src + (size_t)m * lda;
  v16h a;
#pragma unroll
  for (int i = 0; i < 8; ++i) a[i] = p[kh + i];
#pragma unroll
  for (int i = 0; i < 8; ++i) a[8 + i] = p[16 + kh + i];
  return a;
}

// B-matrix 32x16 f16 fragment from *transposed* (N-major) src: srcT[n][k],
// stride ldt halves. lanes 0-15: N=lane, K {0..7,16..23} ; lanes 16-31:
// N=lane-16, K {8..15,24..31}.  Two 8-half (16B) reads -> 2x ds_load_b128.
__device__ __forceinline__ v16h load_bt(const half_t* __restrict__ srcT,
                                        int ldt, int lane) {
  int n  = lane & 15;
  int kb = (lane < 16) ? 0 : 8;
  const half_t* p = srcT + (size_t)n * ldt + kb;
  v16h b;
#pragma unroll
  for (int i = 0; i < 8; ++i) b[i] = p[i];
#pragma unroll
  for (int i = 0; i < 8; ++i) b[8 + i] = p[16 + i];
  return b;
}

// ---------------------------------------------------------------------------
// Async global->LDS copy of 32 contiguous bytes (16 f16), CDNA5 TDM-lite path.
// INST_OFFSET is added to both the LDS and global addresses per the ISA, so
// the second b128 reuses the same address registers with offset:16.
// ---------------------------------------------------------------------------
__device__ __forceinline__ void async_copy16(const half_t* __restrict__ g,
                                             half_t* l) {
  unsigned laddr = (unsigned)(size_t)l;   // low 32 bits = LDS byte offset
  asm volatile("global_load_async_to_lds_b128 %0, %1, off"
               :: "v"(laddr), "v"(g) : "memory");
  asm volatile("global_load_async_to_lds_b128 %0, %1, off offset:16"
               :: "v"(laddr), "v"(g) : "memory");
}
__device__ __forceinline__ void wait_async_all() {
  asm volatile("s_wait_asynccnt 0x0" ::: "memory");
}

// ---------------------------------------------------------------------------
// Problem constants
// ---------------------------------------------------------------------------
#define BB      4
#define CC      512
#define HH      64
#define WW      64
#define WS      32
#define NWIN    16
#define NTOK    1024
#define NHEADS  8
#define DHEAD   64
#define GROUPS  32
#define GCH     16
#define GELEMS  (GCH * HH * WW)

// ---------------------------------------------------------------------------
// Kernel 1: GroupNorm statistics (mean, rstd) per (b, group)
// ---------------------------------------------------------------------------
__global__ __launch_bounds__(256) void gn_stats(const float* __restrict__ x,
                                                float* __restrict__ stats) {
  int bg = blockIdx.x;
  int b = bg >> 5, g = bg & 31;
  const float* src = x + ((size_t)b * CC + (size_t)g * GCH) * (HH * WW);
  float s = 0.f, ss = 0.f;
  for (int i = threadIdx.x; i < GELEMS; i += 256) {
    float v = src[i];
    s += v; ss += v * v;
  }
  __shared__ float rs[256], rq[256];
  int tid = threadIdx.x;
  rs[tid] = s; rq[tid] = ss;
  __syncthreads();
  for (int st = 128; st > 0; st >>= 1) {
    if (tid < st) { rs[tid] += rs[tid + st]; rq[tid] += rq[tid + st]; }
    __syncthreads();
  }
  if (tid == 0) {
    float mean = rs[0] * (1.0f / GELEMS);
    float var  = rq[0] * (1.0f / GELEMS) - mean * mean;
    stats[bg * 2 + 0] = mean;
    stats[bg * 2 + 1] = rsqrtf(var + 1e-6f);
  }
}

// ---------------------------------------------------------------------------
// Kernel 2: convert the four 512x512 weight matrices to f16 (q,k,v,o order)
// ---------------------------------------------------------------------------
__global__ __launch_bounds__(256) void cvt_w(const float* __restrict__ wq,
                                             const float* __restrict__ wk,
                                             const float* __restrict__ wv,
                                             const float* __restrict__ wo,
                                             half_t* __restrict__ w16) {
  int idx = blockIdx.x * 256 + threadIdx.x;
  int m = idx >> 18;
  int r = idx & ((1 << 18) - 1);
  const float* src = (m == 0) ? wq : (m == 1) ? wk : (m == 2) ? wv : wo;
  w16[idx] = (half_t)src[r];
}

// ---------------------------------------------------------------------------
// Kernel 3: GroupNorm apply + window partition, token-major output:
//   hbuf[win][p][c]  (B-operand layout, transposed, for the QKV GEMMs)
// Block = one (win, lh) row: 512 c x 32 lw, transposed through LDS so both
// the f32 reads and the f16 writes stay coalesced.
// ---------------------------------------------------------------------------
__global__ __launch_bounds__(256) void gn_apply(const float* __restrict__ x,
                                                const float* __restrict__ scale,
                                                const float* __restrict__ bias,
                                                const float* __restrict__ stats,
                                                half_t* __restrict__ hbuf) {
  int blk = blockIdx.x;                 // win*32 + lh
  int win = blk >> 5, lh = blk & 31;
  int b = win >> 2, wh = (win >> 1) & 1, wwd = win & 1;
  int h = wh * WS + lh;
  int tid = threadIdx.x;
  const float* xrow = x + ((size_t)b * CC * HH + h) * WW + wwd * WS;
  half_t* dst = hbuf + (size_t)(win * NTOK + lh * WS) * CC;   // [lw][c]

  __shared__ half_t tile[128][36];

  for (int c0 = 0; c0 < CC; c0 += 128) {
    {
      int c = tid >> 1;                 // 0..127
      int lwb = (tid & 1) * 16;
      int cg = c0 + c;
      float mean = stats[(b * GROUPS + (cg >> 4)) * 2 + 0];
      float rstd = stats[(b * GROUPS + (cg >> 4)) * 2 + 1];
      float sc = scale[cg], bi = bias[cg];
      const float* src = xrow + (size_t)cg * (HH * WW) + lwb;
#pragma unroll
      for (int i = 0; i < 16; ++i)
        tile[c][lwb + i] = (half_t)((src[i] - mean) * rstd * sc + bi);
    }
    __syncthreads();
    {
      int lw = tid >> 3;                // 0..31
      int cb = (tid & 7) * 16;
      half_t* d = dst + (size_t)lw * CC + c0 + cb;
#pragma unroll
      for (int i = 0; i < 16; ++i) d[i] = tile[cb + i][lw];
    }
    __syncthreads();
  }
}

// ---------------------------------------------------------------------------
// Kernel 4/6: weight GEMM over all windows.
//   D[o, p_all] = sum_c W[o,c] * Bsrc[win][p][c]   (B source token-major)
// mode 0: Q -> qbuf[wh][n][d]      mode 1: K -> kbuf[wh][n][d]
// mode 2: V -> vbuf[wh][d][n]      mode 3: proj (+bias +residual, f32 out)
// Block = 256 threads = 8 waves, tile 128(M) x 128(N); wave = 64x32.
// A staged row-major, B staged N-major -> all frags are ds_load_b128 pairs,
// all staging via global_load_async_to_lds_b128.
// ---------------------------------------------------------------------------
__global__ __launch_bounds__(256) void gemm512(
    const half_t* __restrict__ w16,
    const half_t* __restrict__ hbuf,
    const half_t* __restrict__ aobuf,
    const float* __restrict__ bq, const float* __restrict__ bk,
    const float* __restrict__ bv, const float* __restrict__ bo,
    const float* __restrict__ xin,
    half_t* __restrict__ qbuf, half_t* __restrict__ kbuf,
    half_t* __restrict__ vbuf, float* __restrict__ outp, int mode_base) {
  int mode = mode_base + blockIdx.z;
  const half_t* W = w16 + (size_t)mode * CC * CC;
  const float* bias = (mode == 0) ? bq : (mode == 1) ? bk
                    : (mode == 2) ? bv : bo;
  const half_t* Bglob = (mode < 3) ? hbuf : aobuf;

  int mtile = blockIdx.y;
  int ntile = blockIdx.x;
  int tid = threadIdx.x;
  int lane = tid & 31;
  int wid = tid >> 5;
  int wm = wid >> 2;                    // 0..1
  int wn = wid & 3;                     // 0..3

  int n0 = ntile * 128;
  int win = n0 >> 10;
  int p0 = n0 & 1023;
  const half_t* Bsrc = Bglob + (size_t)win * NTOK * CC;   // [p][c]

  __shared__ half_t As[128][40];        // 128 rows(M) x 32(K) +8 pad
  __shared__ half_t BshT[128][40];      // 128 rows(N) x 32(K) +8 pad

  v8f acc[4][2] = {};

  int r = tid >> 1;                     // 0..127
  int cb = (tid & 1) * 16;              // 0 / 16

  for (int k0 = 0; k0 < CC; k0 += 32) {
    async_copy16(W + (size_t)(mtile * 128 + r) * CC + k0 + cb, &As[r][cb]);
    async_copy16(Bsrc + (size_t)(p0 + r) * CC + k0 + cb, &BshT[r][cb]);
    wait_async_all();
    __syncthreads();

    v16h bfrag[2];
#pragma unroll
    for (int ni = 0; ni < 2; ++ni)
      bfrag[ni] = load_bt(&BshT[wn * 32 + ni * 16][0], 40, lane);
#pragma unroll
    for (int mi = 0; mi < 4; ++mi) {
      v16h afrag = load_a(&As[wm * 64 + mi * 16][0], 40, lane);
#pragma unroll
      for (int ni = 0; ni < 2; ++ni)
        acc[mi][ni] = wmma_f16(afrag, bfrag[ni], acc[mi][ni]);
    }
    __syncthreads();
  }

  // ---- epilogue ----
  int rbase = (lane < 16) ? 0 : 8;
  int ncol = lane & 15;
#pragma unroll
  for (int mi = 0; mi < 4; ++mi) {
    int o_base = mtile * 128 + wm * 64 + mi * 16 + rbase;
#pragma unroll
    for (int ni = 0; ni < 2; ++ni) {
      int p = p0 + wn * 32 + ni * 16 + ncol;
#pragma unroll
      for (int i = 0; i < 8; ++i) {
        int o = o_base + i;
        float val = acc[mi][ni][i] + bias[o];
        if (mode < 3) {
          int head = o >> 6, dd = o & 63;
          size_t wh = (size_t)(win * NHEADS + head);
          half_t hv = (half_t)val;
          if (mode == 0)      qbuf[(wh * NTOK + p) * DHEAD + dd] = hv;
          else if (mode == 1) kbuf[(wh * NTOK + p) * DHEAD + dd] = hv;
          else                vbuf[(wh * DHEAD + dd) * NTOK + p] = hv;
        } else {
          int lh = p >> 5, lw = p & 31;
          int b = win >> 2, wh2 = (win >> 1) & 1, wwd = win & 1;
          int h = wh2 * WS + lh, w = wwd * WS + lw;
          size_t oidx = (((size_t)b * CC + o) * HH + h) * WW + w;
          outp[oidx] = xin[oidx] + val;
        }
      }
    }
  }
}

// ---------------------------------------------------------------------------
// Kernel 5: flash attention per (win, head). n=1024, d=64, scale=1/8.
// Q,K global: [n][d]; V global: [d][n]. Block = 8 waves x 16 query rows.
// K chunk staged as KsT[key][d] (QK^T B-operand transposed), V chunk as
// VsT[d][key] (PV B-operand transposed) -> contiguous async staging AND
// contiguous fragment loads. Output written token-major to aobuf[win][p][c].
// ---------------------------------------------------------------------------
__global__ __launch_bounds__(256) void attn_kernel(
    const half_t* __restrict__ qbuf, const half_t* __restrict__ kbuf,
    const half_t* __restrict__ vbuf, half_t* __restrict__ aobuf) {
  int pair = blockIdx.y;                // win*8 + head
  int qtile = blockIdx.x;               // 0..7
  int tid = threadIdx.x;
  int lane = tid & 31;
  int wid = tid >> 5;

  const half_t* Q = qbuf + (size_t)pair * NTOK * DHEAD;
  const half_t* K = kbuf + (size_t)pair * NTOK * DHEAD;
  const half_t* V = vbuf + (size_t)pair * DHEAD * NTOK;

  __shared__ half_t KsT[64][72];        // [key][d]
  __shared__ half_t VsT[64][72];        // [d][key]
  __shared__ half_t Ps[8][16][72];      // per-wave P tile (16 rows x 64 keys)

  int qrow = qtile * 128 + wid * 16;
  v16h qa[2];
  qa[0] = load_a(Q + (size_t)qrow * DHEAD + 0, DHEAD, lane);
  qa[1] = load_a(Q + (size_t)qrow * DHEAD + 32, DHEAD, lane);

  v8f o_acc[4] = {};
  float m_run[8], l_run[8];
#pragma unroll
  for (int i = 0; i < 8; ++i) { m_run[i] = -1e30f; l_run[i] = 0.f; }

  int sr = tid >> 2;                    // 0..63
  int scb = (tid & 3) * 16;             // 0..48

  for (int kb = 0; kb < NTOK; kb += 64) {
    async_copy16(K + (size_t)(kb + sr) * DHEAD + scb, &KsT[sr][scb]);
    async_copy16(V + (size_t)sr * NTOK + kb + scb, &VsT[sr][scb]);
    wait_async_all();
    __syncthreads();

    // S = Q * K^T (16 x 64 per wave)
    v8f s[4] = {};
#pragma unroll
    for (int ks = 0; ks < 2; ++ks) {
#pragma unroll
      for (int j = 0; j < 4; ++j) {
        v16h bf = load_bt(&KsT[j * 16][ks * 32], 72, lane);
        s[j] = wmma_f16(qa[ks], bf, s[j]);
      }
    }
#pragma unroll
    for (int j = 0; j < 4; ++j)
#pragma unroll
      for (int i = 0; i < 8; ++i) s[j][i] *= 0.125f;

    // chunk row-max across the 16 lanes of each half-wave
    float mx[8];
#pragma unroll
    for (int i = 0; i < 8; ++i) {
      float m0 = fmaxf(fmaxf(s[0][i], s[1][i]), fmaxf(s[2][i], s[3][i]));
      m0 = fmaxf(m0, __shfl_xor(m0, 1, 32));
      m0 = fmaxf(m0, __shfl_xor(m0, 2, 32));
      m0 = fmaxf(m0, __shfl_xor(m0, 4, 32));
      m0 = fmaxf(m0, __shfl_xor(m0, 8, 32));
      mx[i] = m0;
    }
    float corr[8];
#pragma unroll
    for (int i = 0; i < 8; ++i) {
      float mnew = fmaxf(m_run[i], mx[i]);
      corr[i] = __expf(m_run[i] - mnew);
      m_run[i] = mnew;
      l_run[i] *= corr[i];
    }
#pragma unroll
    for (int j = 0; j < 4; ++j)
#pragma unroll
      for (int i = 0; i < 8; ++i) o_acc[j][i] *= corr[i];

    float rsum[8];
#pragma unroll
    for (int i = 0; i < 8; ++i) rsum[i] = 0.f;
#pragma unroll
    for (int j = 0; j < 4; ++j)
#pragma unroll
      for (int i = 0; i < 8; ++i) {
        float pv = __expf(s[j][i] - m_run[i]);
        s[j][i] = pv;
        rsum[i] += pv;
      }
#pragma unroll
    for (int i = 0; i < 8; ++i) {
      float r0 = rsum[i];
      r0 += __shfl_xor(r0, 1, 32);
      r0 += __shfl_xor(r0, 2, 32);
      r0 += __shfl_xor(r0, 4, 32);
      r0 += __shfl_xor(r0, 8, 32);
      l_run[i] += r0;
    }

    // P -> LDS (per-wave region; same-wave DS ops stay in order)
    {
      int rb = (lane < 16) ? 0 : 8;
      int nc = lane & 15;
#pragma unroll
      for (int j = 0; j < 4; ++j)
#pragma unroll
        for (int i = 0; i < 8; ++i)
          Ps[wid][rb + i][j * 16 + nc] = (half_t)s[j][i];
    }

    // O += P * V
#pragma unroll
    for (int ks = 0; ks < 2; ++ks) {
      v16h pa = load_a(&Ps[wid][0][ks * 32], 72, lane);
#pragma unroll
      for (int j = 0; j < 4; ++j) {
        v16h bf = load_bt(&VsT[j * 16][ks * 32], 72, lane);
        o_acc[j] = wmma_f16(pa, bf, o_acc[j]);
      }
    }
    __syncthreads();
  }

  // finalize: O /= l, write token-major aobuf[win][p][head*64+dd]
  int winb = pair >> 3, head = pair & 7;
  int rb = (lane < 16) ? 0 : 8;
  int nc = lane & 15;
  half_t* AO = aobuf + (size_t)winb * NTOK * CC + (size_t)head * DHEAD;
#pragma unroll
  for (int j = 0; j < 4; ++j) {
    int dd = j * 16 + nc;
#pragma unroll
    for (int i = 0; i < 8; ++i) {
      int p = qrow + rb + i;
      AO[(size_t)p * CC + dd] = (half_t)(o_acc[j][i] / l_run[i]);
    }
  }
}

// ---------------------------------------------------------------------------
// Launch
// ---------------------------------------------------------------------------
extern "C" void kernel_launch(void* const* d_in, const int* in_sizes, int n_in,
                              void* d_out, int out_size, void* d_ws,
                              size_t ws_size, hipStream_t stream) {
  (void)in_sizes; (void)n_in; (void)out_size; (void)ws_size;
  const float* x     = (const float*)d_in[0];
  const float* nsc   = (const float*)d_in[1];
  const float* nbi   = (const float*)d_in[2];
  const float* wq    = (const float*)d_in[3];
  const float* bq    = (const float*)d_in[4];
  const float* wk    = (const float*)d_in[5];
  const float* bk    = (const float*)d_in[6];
  const float* wv    = (const float*)d_in[7];
  const float* bv    = (const float*)d_in[8];
  const float* wo    = (const float*)d_in[9];
  const float* bo    = (const float*)d_in[10];
  float* out         = (float*)d_out;

  char* ws = (char*)d_ws;
  size_t off = 0;
  auto carve = [&](size_t bytes) {
    size_t o = off;
    off += (bytes + 255) & ~(size_t)255;
    return o;
  };
  const size_t half_buf = (size_t)NWIN * CC * NTOK * sizeof(half_t);
  float*  stats = (float*)(ws + carve(BB * GROUPS * 2 * sizeof(float)));
  half_t* w16   = (half_t*)(ws + carve((size_t)4 * CC * CC * sizeof(half_t)));
  half_t* hbuf  = (half_t*)(ws + carve(half_buf));
  half_t* qbuf  = (half_t*)(ws + carve(half_buf));
  half_t* kbuf  = (half_t*)(ws + carve(half_buf));
  half_t* vbuf  = (half_t*)(ws + carve(half_buf));
  half_t* aobuf = (half_t*)(ws + carve(half_buf));

  gn_stats<<<BB * GROUPS, 256, 0, stream>>>(x, stats);
  cvt_w<<<(4 * CC * CC) / 256, 256, 0, stream>>>(wq, wk, wv, wo, w16);
  gn_apply<<<NWIN * WS, 256, 0, stream>>>(x, nsc, nbi, stats, hbuf);
  gemm512<<<dim3(128, 4, 3), 256, 0, stream>>>(w16, hbuf, aobuf, bq, bk, bv,
                                               bo, x, qbuf, kbuf, vbuf, out,
                                               0);
  attn_kernel<<<dim3(8, 128), 256, 0, stream>>>(qbuf, kbuf, vbuf, aobuf);
  gemm512<<<dim3(128, 4, 1), 256, 0, stream>>>(w16, hbuf, aobuf, bq, bk, bv,
                                               bo, x, qbuf, kbuf, vbuf, out,
                                               3);
}
